// LLamaAttentionBlock_58634893525391
// MI455X (gfx1250) — compile-verified, matrix operand order checked by reading, not run on hardware
//
#include <hip/hip_runtime.h>

// ---------------- problem constants (from reference) ----------------
#define Bdim 32
#define Sdim 16
#define Ddim 4096
#define Hdim 32
#define HDdim 128
#define MAXLEN 2048
#define CPOS 1024
#define Lvalid (CPOS + Sdim)      // 1040
#define Lpad 1056                 // 33 * 32, padded key length
#define ROWS (Bdim * Sdim)        // 512 rows for projections

typedef __attribute__((ext_vector_type(16))) __bf16 v16bf;
typedef __attribute__((ext_vector_type(8)))  __bf16 v8bf;
typedef __attribute__((ext_vector_type(8)))  float  v8f;

// ---------------- workspace layout (bytes) ----------------
#define WT_OFF  ((size_t)0)                              // 3 x 4096x4096 bf16 (transposed)
#define WT_SZ   ((size_t)3 * Ddim * Ddim * 2)            // 100,663,296
#define XB_OFF  (WT_OFF + WT_SZ)                         // 3 x 512x4096 bf16
#define XB_SZ   ((size_t)3 * ROWS * Ddim * 2)            // 12,582,912
#define Y_OFF   (XB_OFF + XB_SZ)                         // 3 x 512x4096 f32 (raw proj)
#define Y_SZ    ((size_t)3 * ROWS * Ddim * 4)            // 25,165,824
#define QB_OFF  (Y_OFF + Y_SZ)                           // q bf16 (B,H,S,HD)
#define QB_SZ   ((size_t)Bdim * Hdim * Sdim * HDdim * 2) // 4,194,304

static __device__ inline v16bf cat16(v8bf lo, v8bf hi) {
    return __builtin_shufflevector(lo, hi, 0,1,2,3,4,5,6,7,8,9,10,11,12,13,14,15);
}

// ---------------- f32 -> bf16 elementwise ----------------
__global__ void cvt_x_kernel(const float* __restrict__ x, __bf16* __restrict__ xb, int n) {
    int i = blockIdx.x * blockDim.x + threadIdx.x;
    if (i < n) xb[i] = (__bf16)x[i];
}

// ---------------- W (k,n) f32 -> Wt (n,k) bf16, LDS-tiled transpose ----------------
__global__ void cvt_wT_kernel(const float* __restrict__ W, __bf16* __restrict__ Wt) {
    __shared__ float tile[32][33];
    int kbase = blockIdx.x * 32, nbase = blockIdx.y * 32;
    int tx = threadIdx.x, ty = threadIdx.y; // 32 x 8
#pragma unroll
    for (int i = 0; i < 4; ++i)
        tile[ty + 8 * i][tx] = W[(size_t)(kbase + ty + 8 * i) * Ddim + nbase + tx];
    __syncthreads();
#pragma unroll
    for (int i = 0; i < 4; ++i)
        Wt[(size_t)(nbase + ty + 8 * i) * Ddim + kbase + tx] = (__bf16)tile[tx][ty + 8 * i];
}

// ---------------- projection GEMM: Y(512x4096) = Xb(512x4096) @ Wt^T ----------------
// block = 256 (8 waves); each wave computes a 32x32 tile with 2x2 register blocking
// (4 accumulators, A/B fragment reuse -> 4 wmma per 8 b128 loads).
// grid = (Ddim/256 = 16 n-groups, ROWS/32 = 16 m-tiles)
__global__ void proj_kernel(const __bf16* __restrict__ Xb, const __bf16* __restrict__ Wt,
                            float* __restrict__ Y) {
    int wave = threadIdx.x >> 5;
    int lane = threadIdx.x & 31;
    int m    = lane & 15;
    int kh   = lane >> 4;
    int m_base = blockIdx.y * 32;
    int n_base = blockIdx.x * 256 + wave * 32;

    const __bf16* arow0 = Xb + (size_t)(m_base + m) * Ddim;
    const __bf16* arow1 = Xb + (size_t)(m_base + 16 + m) * Ddim;
    const __bf16* brow0 = Wt + (size_t)(n_base + m) * Ddim;       // n = lane&15
    const __bf16* brow1 = Wt + (size_t)(n_base + 16 + m) * Ddim;

    v8f acc00 = {}, acc01 = {}, acc10 = {}, acc11 = {};
#pragma unroll 2
    for (int kk = 0; kk < Ddim / 32; ++kk) {
        int k0 = kk * 32 + 8 * kh;
        v16bf a0 = cat16(*(const v8bf*)(arow0 + k0), *(const v8bf*)(arow0 + k0 + 16));
        v16bf a1 = cat16(*(const v8bf*)(arow1 + k0), *(const v8bf*)(arow1 + k0 + 16));
        v16bf b0 = cat16(*(const v8bf*)(brow0 + k0), *(const v8bf*)(brow0 + k0 + 16));
        v16bf b1 = cat16(*(const v8bf*)(brow1 + k0), *(const v8bf*)(brow1 + k0 + 16));
        acc00 = __builtin_amdgcn_wmma_f32_16x16x32_bf16(false, a0, false, b0, (short)0, acc00, false, false);
        acc01 = __builtin_amdgcn_wmma_f32_16x16x32_bf16(false, a0, false, b1, (short)0, acc01, false, false);
        acc10 = __builtin_amdgcn_wmma_f32_16x16x32_bf16(false, a1, false, b0, (short)0, acc10, false, false);
        acc11 = __builtin_amdgcn_wmma_f32_16x16x32_bf16(false, a1, false, b1, (short)0, acc11, false, false);
    }
#pragma unroll
    for (int r = 0; r < 8; ++r) {
        int row0 = m_base + r + 8 * kh;
        int row1 = row0 + 16;
        int col0 = n_base + m;
        int col1 = col0 + 16;
        Y[(size_t)row0 * Ddim + col0] = acc00[r];
        Y[(size_t)row0 * Ddim + col1] = acc01[r];
        Y[(size_t)row1 * Ddim + col0] = acc10[r];
        Y[(size_t)row1 * Ddim + col1] = acc11[r];
    }
}

// ---------------- RoPE + scatter into caches / q buffer ----------------
// grid = (8192, 3); p=0: q->qb (bf16, b,h,s,d); p=1: k->Kc slice; p=2: v->Vc slice
__global__ void rope_scatter_kernel(const float* __restrict__ Y, const float* __restrict__ emb,
                                    float* __restrict__ kc, float* __restrict__ vc,
                                    __bf16* __restrict__ qb) {
    int p   = blockIdx.y;
    int idx = blockIdx.x * blockDim.x + threadIdx.x; // over 512*4096
    int row = idx >> 12, col = idx & (Ddim - 1);
    int bb = row >> 4, s = row & 15;
    int hh = col >> 7, d = col & (HDdim - 1);
    const float* Yp = Y + (size_t)p * ROWS * Ddim;
    float x = Yp[idx];
    float val;
    if (p < 2) {
        float c = emb[s * HDdim + d];
        float sn = emb[Sdim * HDdim + s * HDdim + d];
        float other = (d < 64) ? Yp[idx + 64] : Yp[idx - 64];
        val = (d < 64) ? (x * c - other * sn) : (x * c + other * sn);
    } else {
        val = x;
    }
    if (p == 0) {
        qb[(((size_t)bb * Hdim + hh) * Sdim + s) * HDdim + d] = (__bf16)val;
    } else {
        float* dst = (p == 1) ? kc : vc;
        dst[(((size_t)bb * Hdim + hh) * MAXLEN + (CPOS + s)) * HDdim + d] = val;
    }
}

// ---------------- fused attention: one workgroup per (b,h) ----------------
// LDS: sc[16][1056] f32 | pb[16][1056] bf16 | red[2][16][16] f32
#define SC_BYTES   ((size_t)16 * Lpad * 4)   // 67584
#define PB_BYTES   ((size_t)16 * Lpad * 2)   // 33792
#define RED_BYTES  ((size_t)2 * 16 * 16 * 4) // 2048
#define ATTN_LDS   (SC_BYTES + PB_BYTES + RED_BYTES)

__global__ void attn_kernel(const __bf16* __restrict__ qb, const float* __restrict__ kc,
                            const float* __restrict__ vc, float* __restrict__ out) {
    extern __shared__ char smem[];
    float*  sc   = (float*)smem;
    __bf16* pb   = (__bf16*)(smem + SC_BYTES);
    float*  red0 = (float*)(smem + SC_BYTES + PB_BYTES);
    float*  red1 = red0 + 256;

    int bh = blockIdx.x;
    int b = bh >> 5, h = bh & 31;
    const __bf16* qbh  = qb + (size_t)bh * Sdim * HDdim;
    const float*  keys = kc + (size_t)bh * MAXLEN * HDdim;
    const float*  vals = vc + (size_t)bh * MAXLEN * HDdim;

    int wave = threadIdx.x >> 5;
    int lane = threadIdx.x & 31;
    int m    = lane & 15;
    int kh   = lane >> 4;
    const float scale = 0.08838834764831845f; // 1/sqrt(128)

    // Phase 1: scores = q @ K^T (tiles of 16 keys per wave)
    for (int t = wave; t < Lpad / 16; t += 8) {
        int key = t * 16 + m; // B-fragment column = key index
        const __bf16* qrow = qbh + (size_t)m * HDdim;
        const float*  krow = keys + (size_t)key * HDdim;
        v8f c = {};
#pragma unroll
        for (int kk = 0; kk < HDdim / 32; ++kk) {
            int d0 = kk * 32 + 8 * kh;
            v16bf a = cat16(*(const v8bf*)(qrow + d0), *(const v8bf*)(qrow + d0 + 16));
            v16bf bfrag;
#pragma unroll
            for (int i = 0; i < 8; ++i) {
                bfrag[i]     = (__bf16)krow[d0 + i];
                bfrag[i + 8] = (__bf16)krow[d0 + 16 + i];
            }
            c = __builtin_amdgcn_wmma_f32_16x16x32_bf16(false, a, false, bfrag, (short)0, c, false, false);
        }
#pragma unroll
        for (int r = 0; r < 8; ++r) {
            int srow = r + 8 * kh;                 // query row
            float v = c[r] * scale;
            if (key > CPOS + srow) v = -1e30f;     // causal mask (also kills padding cols)
            sc[srow * Lpad + key] = v;
        }
    }
    __syncthreads();

    // Phase 2: softmax over each of 16 rows (16 threads per row, 66 cols each)
    {
        int row = threadIdx.x >> 4;
        int seg = threadIdx.x & 15;
        int c0 = seg * (Lpad / 16), c1 = c0 + (Lpad / 16);
        float mx = -3.4e38f;
        for (int c = c0; c < c1; ++c) mx = fmaxf(mx, sc[row * Lpad + c]);
        red0[row * 16 + seg] = mx;
        __syncthreads();
        mx = red0[row * 16];
#pragma unroll
        for (int i = 1; i < 16; ++i) mx = fmaxf(mx, red0[row * 16 + i]);
        float sum = 0.f;
        for (int c = c0; c < c1; ++c) {
            float e = __expf(sc[row * Lpad + c] - mx);
            sc[row * Lpad + c] = e;
            sum += e;
        }
        red1[row * 16 + seg] = sum;
        __syncthreads();
        sum = 0.f;
#pragma unroll
        for (int i = 0; i < 16; ++i) sum += red1[row * 16 + i];
        float inv = 1.0f / sum;
        for (int c = c0; c < c1; ++c)
            pb[row * Lpad + c] = (__bf16)(sc[row * Lpad + c] * inv);
    }
    __syncthreads();

    // Phase 3: out = P @ V (wave w owns 16 d-columns)
    {
        int dcol = wave * 16 + m;                  // B-fragment column = d
        const __bf16* prow = pb + (size_t)m * Lpad; // A row = query m
        v8f acc = {};
        for (int kt = 0; kt < Lpad / 32; ++kt) {
            int k0 = kt * 32 + 8 * kh;
            v16bf a = cat16(*(const v8bf*)(prow + k0), *(const v8bf*)(prow + k0 + 16));
            v16bf bfrag;
#pragma unroll
            for (int i = 0; i < 8; ++i) {
                bfrag[i]     = (__bf16)vals[(size_t)(k0 + i) * HDdim + dcol];
                bfrag[i + 8] = (__bf16)vals[(size_t)(k0 + 16 + i) * HDdim + dcol];
            }
            acc = __builtin_amdgcn_wmma_f32_16x16x32_bf16(false, a, false, bfrag, (short)0, acc, false, false);
        }
#pragma unroll
        for (int r = 0; r < 8; ++r) {
            int srow = r + 8 * kh;
            out[((size_t)(b * Sdim + srow)) * Ddim + h * HDdim + dcol] = acc[r];
        }
    }
}

extern "C" void kernel_launch(void* const* d_in, const int* in_sizes, int n_in,
                              void* d_out, int out_size, void* d_ws, size_t ws_size,
                              hipStream_t stream) {
    const float* Q      = (const float*)d_in[0];
    const float* K      = (const float*)d_in[1];
    const float* V      = (const float*)d_in[2];
    const float* Kcache = (const float*)d_in[3];
    const float* Vcache = (const float*)d_in[4];
    const float* w0     = (const float*)d_in[5];
    const float* w1     = (const float*)d_in[6];
    const float* w2     = (const float*)d_in[7];
    const float* emb    = (const float*)d_in[8];

    float* out_ptr = (float*)d_out;
    float* kc_ptr  = out_ptr + (size_t)Bdim * Sdim * Ddim;               // after out
    float* vc_ptr  = kc_ptr + (size_t)Bdim * Hdim * MAXLEN * HDdim;      // after Kc

    char*   ws = (char*)d_ws;
    __bf16* Wt = (__bf16*)(ws + WT_OFF);
    __bf16* Xb = (__bf16*)(ws + XB_OFF);
    float*  Y  = (float*)(ws + Y_OFF);
    __bf16* qb = (__bf16*)(ws + QB_OFF);

    const size_t cacheBytes = (size_t)Bdim * Hdim * MAXLEN * HDdim * sizeof(float);
    // Bulk cache passthrough (dominant ~4.3 GB of HBM traffic)
    hipMemcpyAsync(kc_ptr, Kcache, cacheBytes, hipMemcpyDeviceToDevice, stream);
    hipMemcpyAsync(vc_ptr, Vcache, cacheBytes, hipMemcpyDeviceToDevice, stream);

    // f32 -> bf16 activations
    const int nx = ROWS * Ddim; // 2,097,152
    cvt_x_kernel<<<nx / 256, 256, 0, stream>>>(Q, Xb, nx);
    cvt_x_kernel<<<nx / 256, 256, 0, stream>>>(K, Xb + (size_t)nx, nx);
    cvt_x_kernel<<<nx / 256, 256, 0, stream>>>(V, Xb + (size_t)2 * nx, nx);

    // f32 -> bf16 transposed weights
    dim3 wtg(Ddim / 32, Ddim / 32), wtb(32, 8);
    cvt_wT_kernel<<<wtg, wtb, 0, stream>>>(w0, Wt);
    cvt_wT_kernel<<<wtg, wtb, 0, stream>>>(w1, Wt + (size_t)Ddim * Ddim);
    cvt_wT_kernel<<<wtg, wtb, 0, stream>>>(w2, Wt + (size_t)2 * Ddim * Ddim);

    // WMMA projections (32x32 tile per wave, 2x2 register blocking)
    dim3 pg(Ddim / 256, ROWS / 32); // (16, 16)
    proj_kernel<<<pg, 256, 0, stream>>>(Xb, Wt, Y);
    proj_kernel<<<pg, 256, 0, stream>>>(Xb + (size_t)nx, Wt + (size_t)Ddim * Ddim, Y + (size_t)nx);
    proj_kernel<<<pg, 256, 0, stream>>>(Xb + (size_t)2 * nx, Wt + (size_t)2 * Ddim * Ddim, Y + (size_t)2 * nx);

    // RoPE + scatter into caches / q buffer
    rope_scatter_kernel<<<dim3(nx / 256, 3), 256, 0, stream>>>(Y, emb, kc_ptr, vc_ptr, qb);

    // Fused LDS-resident attention
    attn_kernel<<<Bdim * Hdim, 256, (int)ATTN_LDS, stream>>>(qb, kc_ptr, vc_ptr, out_ptr);
}
